// SimpleInterestClock_36103495090262
// MI455X (gfx1250) — matrix operand order repaired and checked by reading, not compiled, not run on hardware
//
#include <hip/hip_runtime.h>

// ---------------------------------------------------------------------------
// SimpleInterestClock fused kernels for MI455X (gfx1250, wave32).
//
// Algebraic collapse: logits = h·(q@k_w) + q·k_b ; score = (attn@h)·(q@v_w) + q·v_b
// -> K/V never materialized (13.4 GFLOP -> <1 GFLOP). The workload becomes an
// L2-resident gather (item_emb = 25.6MB << 192MB L2; ~210MB L2 traffic,
// ~35MB HBM ~ 1.5us @ 23.3TB/s). The remaining real matmul
// (8192x64 @ 64x64, twice) runs on v_wmma_f32_16x16x32_f16, and the gather
// uses GLOBAL_LOAD_ASYNC_TO_LDS_B128 (ASYNCcnt path) for deep MLP without
// VGPR/loadcnt pressure.
// ---------------------------------------------------------------------------

typedef __attribute__((ext_vector_type(16))) _Float16 v16h;
typedef __attribute__((ext_vector_type(8)))  float    v8f;

#define B_SZ   4096
#define L_SEQ  200
#define D_DIM  64
#define H_STR  68   // LDS row stride in floats (pad 64->68 to break bank conflicts)

// ---------------------------------------------------------------------------
// Kernel 0: transpose + convert weights to f16 [k][n] so WMMA B-fragments are
// contiguous 16-half loads per lane.
// ---------------------------------------------------------------------------
__global__ void sic_prep_weights(const float* __restrict__ kw,
                                 const float* __restrict__ vw,
                                 _Float16* __restrict__ kwT,
                                 _Float16* __restrict__ vwT) {
  int i = blockIdx.x * blockDim.x + threadIdx.x;   // 0..4095
  if (i < D_DIM * D_DIM) {
    int n = i >> 6, k = i & 63;
    kwT[k * 64 + n] = (_Float16)kw[i];
    vwT[k * 64 + n] = (_Float16)vw[i];
  }
}

// ---------------------------------------------------------------------------
// Kernel 1: WMMA projection of candidate queries.
//   rows 0..4095    -> q = item_emb[pos_items[r]]
//   rows 4096..8191 -> q = item_emb[neg_items[r-4096]]
//   qk[r] = q @ k_w , qv[r] = q @ v_w , ck[r]=q·k_b, cv[r]=q·v_b
// One wave per 16-row tile; v_wmma_f32_16x16x32_f16, K=64 (2 steps), N=64 (4 tiles).
// A layout: lane&15 = M; el 0..15 = K{c..c+15} (contiguous, same k order as B),
// c=(lane>>4)*16.  B: lane&15 = N; el 0..15 = K{koff..koff+15}, koff=(lane>>4)*16.
// D: VGPR i -> row (lane>>4)*8+i, col lane&15.
// ---------------------------------------------------------------------------
__global__ __launch_bounds__(256)
void sic_qproj_wmma(const int* __restrict__ pos_items,
                    const int* __restrict__ neg_items,
                    const float* __restrict__ item_emb,
                    const _Float16* __restrict__ kwT,
                    const _Float16* __restrict__ vwT,
                    const float* __restrict__ k_b,
                    const float* __restrict__ v_b,
                    float* __restrict__ qk, float* __restrict__ qv,
                    float* __restrict__ ck, float* __restrict__ cv) {
  const int wave = threadIdx.x >> 5;
  const int lane = threadIdx.x & 31;
  const int tile = blockIdx.x * 8 + wave;          // 512 tiles total
  const int r0   = tile * 16;

  const int mrow = r0 + (lane & 15);
  const int idx  = (mrow < B_SZ) ? pos_items[mrow] : neg_items[mrow - B_SZ];
  const float* q = item_emb + (long long)idx * D_DIM;
  const int c = (lane >> 4) * 16;
  v16h a[2];
#pragma unroll
  for (int s = 0; s < 2; ++s) {
#pragma unroll
    for (int j = 0; j < 8; ++j) {
      a[s][j]     = (_Float16)q[s * 32 + c + j];
      a[s][8 + j] = (_Float16)q[s * 32 + c + 8 + j];
    }
  }

  {
    const float* bias = (lane < 16) ? k_b : v_b;
    float s = 0.f;
#pragma unroll
    for (int j = 0; j < D_DIM; ++j) s += q[j] * bias[j];
    if (lane < 16) ck[mrow] = s; else cv[mrow] = s;
  }

  const int col  = lane & 15;
  const int koff = (lane >> 4) * 16;
#pragma unroll
  for (int ct = 0; ct < 4; ++ct) {
    const int k = ct * 16 + col;
    v16h bk[2], bv[2];
#pragma unroll
    for (int s = 0; s < 2; ++s) {
      const _Float16* pk = kwT + k * 64 + s * 32 + koff;
      const _Float16* pv = vwT + k * 64 + s * 32 + koff;
#pragma unroll
      for (int j = 0; j < 16; ++j) { bk[s][j] = pk[j]; bv[s][j] = pv[j]; }
    }
    v8f ak = {}, av = {};
#pragma unroll
    for (int s = 0; s < 2; ++s) {
      ak = __builtin_amdgcn_wmma_f32_16x16x32_f16(false, a[s], false, bk[s],
                                                  (short)0, ak, false, false);
      av = __builtin_amdgcn_wmma_f32_16x16x32_f16(false, a[s], false, bv[s],
                                                  (short)0, av, false, false);
    }
    const int orow = r0 + (lane >> 4) * 8;
#pragma unroll
    for (int i = 0; i < 8; ++i) {
      qk[(long long)(orow + i) * 64 + k] = ak[i];
      qv[(long long)(orow + i) * 64 + k] = av[i];
    }
  }
}

// ---------------------------------------------------------------------------
// Kernel 2: per batch-row fused gather + clock-encode + dual masked-softmax
// attention + scores.  1 block (256 thr = 8 wave32) per batch row.
// item_emb gather uses GLOBAL_LOAD_ASYNC_TO_LDS_B128 (ASYNCcnt).
// ---------------------------------------------------------------------------
__global__ __launch_bounds__(256)
void sic_attend(const int* __restrict__ items, const int* __restrict__ dts,
                const unsigned char* __restrict__ mask,
                const float* __restrict__ item_emb,
                const float* __restrict__ dt_emb,
                const float* __restrict__ clock_w,
                const float* __restrict__ clock_b,
                const float* __restrict__ raw_tau,
                const float* __restrict__ qk, const float* __restrict__ qv,
                const float* __restrict__ ck, const float* __restrict__ cv,
                float* __restrict__ out) {
  __shared__ float h[L_SEQ][H_STR];                // 54,400 B
  __shared__ float qkp[64], qkn[64], qvp[64], qvn[64];
  __shared__ float lgat[2][L_SEQ];                 // logits -> exp -> attn, in place
  __shared__ float stat[4];                        // maxp, maxn, sump, sumn
  // phase-0 staging and phase-3 reduction have disjoint lifetimes -> union
  __shared__ union {
    struct {
      int   it[L_SEQ]; int dt[L_SEQ];
      float sn[L_SEQ]; float cs[L_SEQ];
      float cw[128];   float cb[64];
    } p0;                                          // 3,968 B
    struct {
      float red[2][4][64];                         // 2,048 B
      float hb[2][64];                             //   512 B
    } p1;
  } u;

  const int b = blockIdx.x;
  const int t = threadIdx.x;

  // ---- stage q-projection vectors (256 threads = 4 x 64 loads exactly) ----
  if      (t < 64)  qkp[t]       = qk[(long long)b * 64 + t];
  else if (t < 128) qkn[t - 64]  = qk[(long long)(B_SZ + b) * 64 + (t - 64)];
  else if (t < 192) qvp[t - 128] = qv[(long long)b * 64 + (t - 128)];
  else              qvn[t - 192] = qv[(long long)(B_SZ + b) * 64 + (t - 192)];

  // ---- phase 0: per-row indices + sincos, clock params -> LDS ----
  const float w = 6.28318530717958647692f / 127.0f;   // 2*pi/(NUM_DT-1)
  if (t < L_SEQ) {
    const int it = items[b * L_SEQ + t];
    const int dt = dts[b * L_SEQ + t];
    u.p0.it[t] = it;
    u.p0.dt[t] = dt;
    float s, co;
    __sincosf(w * (float)dt, &s, &co);
    u.p0.sn[t] = s;
    u.p0.cs[t] = co;
  }
  if (t < 128)                u.p0.cw[t]       = clock_w[t];
  else if (t < 192)           u.p0.cb[t - 128] = clock_b[t - 128];
  __syncthreads();

  // ---- phase 1: async gather of item_emb rows into h (16B per lane-op) ----
  for (int e = t; e < L_SEQ * 16; e += 256) {
    const int l  = e >> 4;
    const int dq = e & 15;                         // float4 index within row
    const float*   g    = item_emb + (long long)u.p0.it[l] * D_DIM + dq * 4;
    const unsigned lofs = (unsigned)(size_t)&h[l][dq * 4];   // LDS byte address
    asm volatile("global_load_async_to_lds_b128 %0, %1, off"
                 :: "v"(lofs), "v"(g) : "memory");
  }
  asm volatile("s_wait_asynccnt 0x0" ::: "memory");
  __syncthreads();   // every wave waited its own ASYNCcnt -> all copies visible

  // ---- phase 2: in-place add of dt_emb + clock encoding (float4) ----
  for (int e = t; e < L_SEQ * 16; e += 256) {
    const int l  = e >> 4;
    const int dq = e & 15;
    const int d  = dq * 4;
    const int dt = u.p0.dt[l];
    const float  sn = u.p0.sn[l], cs = u.p0.cs[l];
    float4 hv = *(float4*)&h[l][d];
    const float4 de = *(const float4*)(dt_emb + dt * 64 + d);
    hv.x += de.x + sn * u.p0.cw[2 * d + 0] + cs * u.p0.cw[2 * d + 1] + u.p0.cb[d + 0];
    hv.y += de.y + sn * u.p0.cw[2 * d + 2] + cs * u.p0.cw[2 * d + 3] + u.p0.cb[d + 1];
    hv.z += de.z + sn * u.p0.cw[2 * d + 4] + cs * u.p0.cw[2 * d + 5] + u.p0.cb[d + 2];
    hv.w += de.w + sn * u.p0.cw[2 * d + 6] + cs * u.p0.cw[2 * d + 7] + u.p0.cb[d + 3];
    *(float4*)&h[l][d] = hv;
  }
  __syncthreads();

  const float tau     = log1pf(__expf(raw_tau[0])) + 1e-6f;
  const float inv_tau = 1.0f / tau;

  // ---- logits (both candidates in one pass over h) ----
  if (t < L_SEQ) {
    float dp = 0.f, dn = 0.f;
#pragma unroll
    for (int d = 0; d < 64; d += 4) {
      const float4 hv = *(const float4*)&h[t][d];
      dp += hv.x * qkp[d] + hv.y * qkp[d + 1] + hv.z * qkp[d + 2] + hv.w * qkp[d + 3];
      dn += hv.x * qkn[d] + hv.y * qkn[d + 1] + hv.z * qkn[d + 2] + hv.w * qkn[d + 3];
    }
    const bool m = mask[b * L_SEQ + t] != 0;
    lgat[0][t] = m ? (dp + ck[b])        * inv_tau : -INFINITY;
    lgat[1][t] = m ? (dn + ck[B_SZ + b]) * inv_tau : -INFINITY;
  }
  __syncthreads();

  // ---- row max (wave 0 = pos, wave 1 = neg) ----
  if (t < 64) {
    const int cnd = t >> 5, lane = t & 31;
    float m = -INFINITY;
    for (int i = lane; i < L_SEQ; i += 32) m = fmaxf(m, lgat[cnd][i]);
#pragma unroll
    for (int off = 16; off > 0; off >>= 1) m = fmaxf(m, __shfl_xor(m, off, 32));
    if (lane == 0) stat[cnd] = m;
  }
  __syncthreads();

  // ---- exp (in place; exp(-inf)=0 handles masked) ----
  if (t < L_SEQ) {
    lgat[0][t] = __expf(lgat[0][t] - stat[0]);
    lgat[1][t] = __expf(lgat[1][t] - stat[1]);
  }
  __syncthreads();

  // ---- sum ----
  if (t < 64) {
    const int cnd = t >> 5, lane = t & 31;
    float s = 0.f;
    for (int i = lane; i < L_SEQ; i += 32) s += lgat[cnd][i];
#pragma unroll
    for (int off = 16; off > 0; off >>= 1) s += __shfl_xor(s, off, 32);
    if (lane == 0) stat[2 + cnd] = s;
  }
  __syncthreads();

  // ---- normalize (in place); emit attn_pos ----
  if (t < L_SEQ) {
    const float ap = lgat[0][t] / stat[2];
    const float an = lgat[1][t] / stat[3];
    lgat[0][t] = ap;
    lgat[1][t] = an;
    out[2 * B_SZ + (long long)b * L_SEQ + t] = ap;
  }
  __syncthreads();

  // ---- hbar = attn @ h  (column-parallel, conflict-free LDS reads) ----
  {
    const int rg = t >> 6, d = t & 63;
    float sp = 0.f, sn = 0.f;
    for (int l = rg; l < L_SEQ; l += 4) {
      const float hv = h[l][d];
      sp += lgat[0][l] * hv;
      sn += lgat[1][l] * hv;
    }
    u.p1.red[0][rg][d] = sp;
    u.p1.red[1][rg][d] = sn;
  }
  __syncthreads();
  if (t < 128) {
    const int cnd = t >> 6, d = t & 63;
    u.p1.hb[cnd][d] = u.p1.red[cnd][0][d] + u.p1.red[cnd][1][d] +
                      u.p1.red[cnd][2][d] + u.p1.red[cnd][3][d];
  }
  __syncthreads();

  // ---- scores: score = hbar·qv + cv ----
  if (t < 64) {
    const int cnd = t >> 5, lane = t & 31;
    const float* qvv = cnd ? qvn : qvp;
    float s = u.p1.hb[cnd][lane] * qvv[lane] + u.p1.hb[cnd][lane + 32] * qvv[lane + 32];
#pragma unroll
    for (int off = 16; off > 0; off >>= 1) s += __shfl_xor(s, off, 32);
    if (lane == 0) out[cnd * B_SZ + b] = s + cv[cnd ? (B_SZ + b) : b];
  }
}

// ---------------------------------------------------------------------------
extern "C" void kernel_launch(void* const* d_in, const int* in_sizes, int n_in,
                              void* d_out, int out_size, void* d_ws, size_t ws_size,
                              hipStream_t stream) {
  (void)in_sizes; (void)n_in; (void)out_size; (void)ws_size;
  const int*           items     = (const int*)d_in[0];
  const int*           dts       = (const int*)d_in[1];
  const unsigned char* mask      = (const unsigned char*)d_in[2];   // jax bool = 1 byte
  const int*           pos_items = (const int*)d_in[3];
  const int*           neg_items = (const int*)d_in[4];
  const float*         item_emb  = (const float*)d_in[5];
  const float*         dt_emb    = (const float*)d_in[6];
  const float*         clock_w   = (const float*)d_in[7];
  const float*         clock_b   = (const float*)d_in[8];
  const float*         k_w       = (const float*)d_in[9];
  const float*         k_b       = (const float*)d_in[10];
  const float*         v_w       = (const float*)d_in[11];
  const float*         v_b       = (const float*)d_in[12];
  const float*         raw_tau   = (const float*)d_in[13];
  float* out = (float*)d_out;

  // workspace carve-out (~4.2 MB)
  char* ws = (char*)d_ws;
  float*    qk  = (float*)ws;     ws += (size_t)2 * B_SZ * 64 * sizeof(float);
  float*    qv  = (float*)ws;     ws += (size_t)2 * B_SZ * 64 * sizeof(float);
  float*    ck  = (float*)ws;     ws += (size_t)2 * B_SZ * sizeof(float);
  float*    cv  = (float*)ws;     ws += (size_t)2 * B_SZ * sizeof(float);
  _Float16* kwT = (_Float16*)ws;  ws += (size_t)64 * 64 * sizeof(_Float16);
  _Float16* vwT = (_Float16*)ws;

  sic_prep_weights<<<16, 256, 0, stream>>>(k_w, v_w, kwT, vwT);
  sic_qproj_wmma<<<64, 256, 0, stream>>>(pos_items, neg_items, item_emb,
                                         kwT, vwT, k_b, v_b, qk, qv, ck, cv);
  sic_attend<<<B_SZ, 256, 0, stream>>>(items, dts, mask, item_emb, dt_emb,
                                       clock_w, clock_b, raw_tau,
                                       qk, qv, ck, cv, out);
}
